// BertSelfAttention_37546604101659
// MI455X (gfx1250) — matrix operand order repaired and checked by reading, compiled, not run on hardware
//
#include <hip/hip_runtime.h>
#include <hip/hip_bf16.h>

// ---------------------------------------------------------------------------
// BERT self-attention forward for MI455X (gfx1250, wave32, WMMA).
//   B=4, S=2048, D=1024, H=16, DH=64
// f32 -> bf16 QKV GEMMs (v_wmma_f32_16x16x32_bf16, f32 accum), then
// flash-attention fused softmax(QK^T)V on bf16 WMMA.
//  - Q,K stored [B,H,S,DH]; V stored TRANSPOSED [B,H,DH,S] by the GEMM
//    epilogue (compile-time strides) so every WMMA B-fragment in attention
//    reads a contiguous K-dimension (32B LDS read -> two ds_load_b128).
//  - Staging loops: exact trip counts (no predication), load/store phases
//    split so multiple global_load_b128 stay in flight.
// ---------------------------------------------------------------------------

#define B_  4
#define S_  2048
#define D_  1024
#define H_  16
#define DH_ 64

typedef __attribute__((ext_vector_type(16))) __bf16 v16bf;
typedef __attribute__((ext_vector_type(8)))  __bf16 v8bf;
typedef __attribute__((ext_vector_type(4)))  __bf16 v4bf;
typedef __attribute__((ext_vector_type(8)))  float  v8f;
typedef __attribute__((ext_vector_type(4)))  float  v4f;

static __device__ __forceinline__ v8f wmma_bf16(const v16bf& a, const v16bf& b, const v8f& c) {
  return __builtin_amdgcn_wmma_f32_16x16x32_bf16(false, a, false, b, (short)0, c, false, false);
}

// Build a 16x32 A-fragment from two contiguous 16-byte chunks (b128 loads).
static __device__ __forceinline__ v16bf ld_frag(const __bf16* p0, const __bf16* p1) {
  v8bf lo = *(const v8bf*)p0;
  v8bf hi = *(const v8bf*)(p0 + 8);
  return __builtin_shufflevector(lo, hi, 0, 1, 2, 3, 4, 5, 6, 7,
                                         8, 9, 10, 11, 12, 13, 14, 15);
}

// ---------------------------------------------------------------------------
// Kernel 1: out = hidden @ W + bias, written bf16 with compile-time strides:
//   element (b*S+s, h*64+dh) -> out[(b*H+h)*S*DH + s*SS + dh*SD]
//   Q/K: <DH,1> -> [B,H,S,DH]      V: <1,S> -> [B,H,DH,S] (pre-transposed)
// Block: 128 threads (4 waves). Tile: 64 rows x 64 cols. K step: 32.
// ---------------------------------------------------------------------------
template <int SS, int SD>
__global__ __launch_bounds__(128)
void qkv_gemm(const float* __restrict__ hidden, const float* __restrict__ W,
              const float* __restrict__ bias, __bf16* __restrict__ out) {
  const int tid  = threadIdx.x;
  const int wave = tid >> 5;
  const int lane = tid & 31;
  const int row0 = blockIdx.x * 64;     // global row = b*S + s
  const int col0 = blockIdx.y * 64;     // global col in D

  __shared__ __align__(16) __bf16 As[64][32];    // [row][k]       (A layout)
  __shared__ __align__(16) __bf16 Bst[64][32];   // [col][k] (transposed W)

  const int wr = (wave >> 1) * 32;
  const int wc = (wave & 1) * 32;

  const int arow = lane & 15;           // A: row within 16
  const int akb  = (lane >> 4) * 16;    // A: K sub-base per lane half
  const int bcol = lane & 15;           // B: column
  const int bkb  = (lane >> 4) * 16;    // B: K base per lane half
  const int hi   = lane >> 4;
  const int lc   = lane & 15;

  v8f acc[2][2] = {};

  for (int k0 = 0; k0 < D_; k0 += 32) {
    // ---- load phase: 8 independent global_load_b128 in flight
    v4f fa[4], fw[4];
#pragma unroll
    for (int it = 0; it < 4; ++it) {
      const int i = it * 128 + tid;     // A tile 64x32: 8 float4 per row
      const int r = i >> 3, c = (i & 7) * 4;
      fa[it] = *(const v4f*)&hidden[(size_t)(row0 + r) * D_ + (k0 + c)];
    }
#pragma unroll
    for (int it = 0; it < 4; ++it) {
      const int i = it * 128 + tid;     // W tile 32x64: 16 float4 per row
      const int r = i >> 4, c = (i & 15) * 4;
      fw[it] = *(const v4f*)&W[(size_t)(k0 + r) * D_ + (col0 + c)];
    }
    // Speculative prefetch of next A tile (global_prefetch_b8); harmlessly
    // dropped on translation failure past the last tile (ISA TH=0).
    __builtin_prefetch(&hidden[(size_t)(row0 + (tid >> 1)) * D_ + (k0 + 32)], 0, 1);

    // ---- store phase: packed bf16 converts + LDS stores
#pragma unroll
    for (int it = 0; it < 4; ++it) {
      const int i = it * 128 + tid;
      const int r = i >> 3, c = (i & 7) * 4;
      v4bf o;
      o[0] = (__bf16)fa[it][0]; o[1] = (__bf16)fa[it][1];
      o[2] = (__bf16)fa[it][2]; o[3] = (__bf16)fa[it][3];
      *(v4bf*)&As[r][c] = o;
    }
#pragma unroll
    for (int it = 0; it < 4; ++it) {
      const int i = it * 128 + tid;
      const int r = i >> 4, c = (i & 15) * 4;
#pragma unroll
      for (int j = 0; j < 4; ++j) Bst[c + j][r] = (__bf16)fw[it][j];
    }
    __syncthreads();

    v16bf afrag[2], bfrag[2];
#pragma unroll
    for (int m = 0; m < 2; ++m)
      afrag[m] = ld_frag(&As[wr + m * 16 + arow][akb],
                         &As[wr + m * 16 + arow][16 + akb]);
#pragma unroll
    for (int n = 0; n < 2; ++n)
      bfrag[n] = *(const v16bf*)&Bst[wc + n * 16 + bcol][bkb];

#pragma unroll
    for (int m = 0; m < 2; ++m)
#pragma unroll
      for (int n = 0; n < 2; ++n)
        acc[m][n] = wmma_bf16(afrag[m], bfrag[n], acc[m][n]);
    __syncthreads();
  }

  // Epilogue: bias, convert bf16, scatter with compile-time strides
#pragma unroll
  for (int m = 0; m < 2; ++m)
#pragma unroll
    for (int n = 0; n < 2; ++n)
#pragma unroll
      for (int r = 0; r < 8; ++r) {
        const int grow = row0 + wr + m * 16 + r + hi * 8;   // b*S + s
        const int gcol = col0 + wc + n * 16 + lc;           // h*64 + dh
        const float v  = acc[m][n][r] + bias[gcol];
        const int b = grow >> 11, s = grow & (S_ - 1);
        const int h = gcol >> 6, dh = gcol & 63;
        out[((size_t)b * H_ + h) * (S_ * DH_) + (size_t)s * SS + (size_t)dh * SD]
            = (__bf16)v;
      }
}

// ---------------------------------------------------------------------------
// Kernel 2: flash attention. Grid (S/64, H, B), 128 threads (4 waves).
// Each wave owns 16 q rows; KV processed in 64-wide tiles from LDS.
//   Kt[key][dh] : K-dim (dh) contiguous for QK^T B-fragments
//   Vt[dh][kv]  : K-dim (kv) contiguous for P.V  B-fragments
//                 (V arrives pre-transposed [B,H,DH,S] -> plain b128 copies)
// ---------------------------------------------------------------------------
__global__ __launch_bounds__(128)
void attn(const __bf16* __restrict__ Q, const __bf16* __restrict__ K,
          const __bf16* __restrict__ V, const float* __restrict__ mask,
          float* __restrict__ out) {
  const int tid  = threadIdx.x;
  const int wave = tid >> 5;
  const int lane = tid & 31;
  const int q0 = blockIdx.x * 64;
  const int h  = blockIdx.y;
  const int b  = blockIdx.z;

  const size_t head = ((size_t)b * H_ + h) * (S_ * DH_);
  const __bf16* Qh = Q + head;          // [s][dh]
  const __bf16* Kh = K + head;          // [s][dh]
  const __bf16* Vh = V + head;          // [dh][s]  (pre-transposed)

  __shared__ __align__(16) __bf16 Kt[64][DH_];      // [key][dh]
  __shared__ __align__(16) __bf16 Vt[DH_][64];      // [dh][key]
  __shared__ __align__(16) __bf16 Pt[4][16][64];    // per-wave probs scratch

  const int arow = lane & 15;
  const int akb  = (lane >> 4) * 16;
  const int bcol = lane & 15;
  const int bkb  = (lane >> 4) * 16;
  const int hi   = lane >> 4;
  const int lc   = lane & 15;

  // Q A-fragments: 16 q-rows x 64 dh = two 16x32 fragments, loaded once.
  const int qrow = q0 + wave * 16 + arow;
  v16bf qf[2];
#pragma unroll
  for (int ks = 0; ks < 2; ++ks)
    qf[ks] = ld_frag(&Qh[(size_t)qrow * DH_ + ks * 32 + akb],
                     &Qh[(size_t)qrow * DH_ + ks * 32 + 16 + akb]);

  float m_run[8], l_run[8];
  v8f oacc[4] = {};
#pragma unroll
  for (int r = 0; r < 8; ++r) { m_run[r] = -3.0e38f; l_run[r] = 0.0f; }

  const float rscale = 0.125f;          // 1/sqrt(DH)

  for (int kv0 = 0; kv0 < S_; kv0 += 64) {
    // ---- staging: 8 b128 loads in flight, then 8 b128 LDS stores
    v8bf kk[4], vv[4];
#pragma unroll
    for (int it = 0; it < 4; ++it) {
      const int i = it * 128 + tid;     // 8 v8bf per 64-elem row
      const int r = i >> 3, c = (i & 7) * 8;
      kk[it] = *(const v8bf*)&Kh[(size_t)(kv0 + r) * DH_ + c];
      vv[it] = *(const v8bf*)&Vh[(size_t)r * S_ + kv0 + c];
    }
#pragma unroll
    for (int it = 0; it < 4; ++it) {
      const int i = it * 128 + tid;
      const int r = i >> 3, c = (i & 7) * 8;
      *(v8bf*)&Kt[r][c] = kk[it];
      *(v8bf*)&Vt[r][c] = vv[it];
    }
    __syncthreads();

    // ---- scores S = Q K^T : 4 column blocks of 16 keys, 2 K-steps over dh
    v8f sacc[4] = {};
#pragma unroll
    for (int n = 0; n < 4; ++n)
#pragma unroll
      for (int ks = 0; ks < 2; ++ks) {
        v16bf bf = *(const v16bf*)&Kt[n * 16 + bcol][ks * 32 + bkb];
        sacc[n] = wmma_bf16(qf[ks], bf, sacc[n]);
      }

    // ---- scale + mask, then online softmax over this tile
    float rmax[8];
#pragma unroll
    for (int r = 0; r < 8; ++r) rmax[r] = -3.0e38f;
    const float* mrow = &mask[((size_t)b * S_ + (q0 + wave * 16 + hi * 8)) * S_ + kv0];
#pragma unroll
    for (int n = 0; n < 4; ++n)
#pragma unroll
      for (int r = 0; r < 8; ++r) {
        float sv = sacc[n][r] * rscale + mrow[(size_t)r * S_ + n * 16 + lc];
        sacc[n][r] = sv;
        rmax[r] = fmaxf(rmax[r], sv);
      }
#pragma unroll
    for (int off = 1; off < 16; off <<= 1)
#pragma unroll
      for (int r = 0; r < 8; ++r)
        rmax[r] = fmaxf(rmax[r], __shfl_xor(rmax[r], off, 32));

    float rsum[8], scl[8];
#pragma unroll
    for (int r = 0; r < 8; ++r) {
      const float mn = fmaxf(m_run[r], rmax[r]);
      scl[r]   = __expf(m_run[r] - mn);
      m_run[r] = mn;
      rsum[r]  = 0.0f;
    }
#pragma unroll
    for (int n = 0; n < 4; ++n)
#pragma unroll
      for (int r = 0; r < 8; ++r) {
        const float p = __expf(sacc[n][r] - m_run[r]);
        sacc[n][r] = p;
        rsum[r] += p;
      }
#pragma unroll
    for (int off = 1; off < 16; off <<= 1)
#pragma unroll
      for (int r = 0; r < 8; ++r)
        rsum[r] += __shfl_xor(rsum[r], off, 32);
#pragma unroll
    for (int r = 0; r < 8; ++r) l_run[r] = l_run[r] * scl[r] + rsum[r];

    // rescale output accumulators (flash update)
#pragma unroll
    for (int n = 0; n < 4; ++n)
#pragma unroll
      for (int r = 0; r < 8; ++r) oacc[n][r] *= scl[r];

    // ---- probs: C-layout -> LDS -> A-fragment relayout (bf16)
#pragma unroll
    for (int n = 0; n < 4; ++n)
#pragma unroll
      for (int r = 0; r < 8; ++r)
        Pt[wave][r + hi * 8][n * 16 + lc] = (__bf16)sacc[n][r];
    __syncthreads();

    // ---- O += P V : two K-steps over kv, 4 column blocks of dh
#pragma unroll
    for (int ks = 0; ks < 2; ++ks) {
      v16bf pf = ld_frag(&Pt[wave][arow][ks * 32 + akb],
                         &Pt[wave][arow][ks * 32 + 16 + akb]);
#pragma unroll
      for (int n = 0; n < 4; ++n) {
        v16bf vf = *(const v16bf*)&Vt[n * 16 + lc][ks * 32 + bkb];
        oacc[n] = wmma_bf16(pf, vf, oacc[n]);
      }
    }
    __syncthreads();   // protect Kt/Vt/Pt before next tile
  }

  // ---- epilogue: normalize (8 rcp + 32 mul), write ctx [B,S,H*DH] f32
  float rl[8];
#pragma unroll
  for (int r = 0; r < 8; ++r) rl[r] = 1.0f / l_run[r];
#pragma unroll
  for (int n = 0; n < 4; ++n)
#pragma unroll
    for (int r = 0; r < 8; ++r) {
      const int s = q0 + wave * 16 + r + hi * 8;
      out[((size_t)b * S_ + s) * D_ + h * DH_ + n * 16 + lc] =
          oacc[n][r] * rl[r];
    }
}

// ---------------------------------------------------------------------------
extern "C" void kernel_launch(void* const* d_in, const int* in_sizes, int n_in,
                              void* d_out, int out_size, void* d_ws, size_t ws_size,
                              hipStream_t stream) {
  const float* hidden = (const float*)d_in[0];
  const float* mask   = (const float*)d_in[1];
  const float* Wq     = (const float*)d_in[2];
  const float* bq     = (const float*)d_in[3];
  const float* Wk     = (const float*)d_in[4];
  const float* bk     = (const float*)d_in[5];
  const float* Wv     = (const float*)d_in[6];
  const float* bv     = (const float*)d_in[7];
  float* out = (float*)d_out;

  const size_t elems = (size_t)B_ * H_ * S_ * DH_;   // 8M bf16 elems per buffer
  __bf16* qbuf = (__bf16*)d_ws;
  __bf16* kbuf = qbuf + elems;
  __bf16* vbuf = kbuf + elems;

  dim3 gg((B_ * S_) / 64, D_ / 64);                  // (128, 16)
  qkv_gemm<DH_, 1><<<gg, 128, 0, stream>>>(hidden, Wq, bq, qbuf); // [B,H,S,DH]
  qkv_gemm<DH_, 1><<<gg, 128, 0, stream>>>(hidden, Wk, bk, kbuf); // [B,H,S,DH]
  qkv_gemm<1, S_><<<gg, 128, 0, stream>>>(hidden, Wv, bv, vbuf);  // [B,H,DH,S]

  dim3 ga(S_ / 64, H_, B_);                          // (32, 16, 4)
  attn<<<ga, 128, 0, stream>>>(qbuf, kbuf, vbuf, mask, out);
}
